// BaselineMoE_33930241638657
// MI455X (gfx1250) — matrix operand order repaired and checked by reading, not compiled, not run on hardware
//
#include <hip/hip_runtime.h>
#include <math.h>

#define TT 4096   // B*T tokens
#define DD 2048   // model dim
#define FF 2048   // hidden dim
#define NE 8      // routed experts
#define NXP 9     // routed + shared
#define TM 32     // tokens per workgroup (two 16-row WMMA subtiles)
#define LDP 2056  // padded LDS row stride (bf16 elems): 4112 B % 256 B = 16 -> conflict-free

typedef __attribute__((ext_vector_type(8)))  float  v8f;
typedef __attribute__((ext_vector_type(8)))  __bf16 v8bf;
typedef __attribute__((ext_vector_type(16))) __bf16 v16bf;

union BF16Frag {
    v16bf v;
    v8bf  h[2];
    uint4 q[2];
};

union BF8Cvt { v8bf b; uint4 q; };

__device__ __forceinline__ unsigned short f2bf(float f) {
    union { __bf16 b; unsigned short u; } cv;
    cv.b = (__bf16)f;                    // hardware cvt, RNE
    return cv.u;
}

__device__ __forceinline__ v8f wmma_bf16(v16bf a, v16bf b, v8f c) {
    return __builtin_amdgcn_wmma_f32_16x16x32_bf16(
        /*neg_a=*/false, a, /*neg_b=*/false, b,
        /*c_mod=*/(short)0, c, /*reuse_a=*/false, /*reuse_b=*/false);
}

// B fragment (32x16 bf16, KxN) from row-major fp32 weight w[outfeat][k].
// Lane L: N=L&15, g=L>>4; halves 0..7 -> K = kbase + 8g + h ; 8..15 -> K = kbase + 16 + 8g + (h-8).
__device__ __forceinline__ v16bf load_b_global(const float* __restrict__ w, int ldw,
                                               int nbase, int kbase, int lane) {
    const int n = lane & 15, g = (lane >> 4) & 1;
    const float* p = w + (size_t)(nbase + n) * ldw + kbase + g * 8;
    BF16Frag fr;
    fr.h[0] = __builtin_convertvector(*(const v8f*)(p),     v8bf);  // v_cvt_pk_bf16_f32 x4
    fr.h[1] = __builtin_convertvector(*(const v8f*)(p + 16), v8bf);
    return fr.v;
}

// A fragment (16x32 bf16) from LDS tile (row stride ldk elements).
// Lane L: M=L&15, g=L>>4. Halves 0..7 -> K=kbase+8g+h ; 8..15 -> K=kbase+16+8g+(h-8).
__device__ __forceinline__ v16bf load_a_lds(const unsigned short* s, int ldk,
                                            int kbase, int lane) {
    const int m = lane & 15, g = (lane >> 4) & 1;
    const unsigned short* p = s + m * ldk + kbase + g * 8;
    BF16Frag fr;
    fr.q[0] = *(const uint4*)(p);        // halves 0..7
    fr.q[1] = *(const uint4*)(p + 16);   // halves 8..15
    return fr.v;
}

// ---------------- Router: sigmoid(x @ W) + top-2 -> dense (T,9) gate table ----
__global__ __launch_bounds__(256)
void router_kernel(const float* __restrict__ x, const float* __restrict__ rw,
                   float* __restrict__ gates) {
    const int lane = threadIdx.x & 31;
    const int wave = threadIdx.x >> 5;
    const int t = blockIdx.x * 8 + wave;
    const float* xr = x + (size_t)t * DD;

    float acc[NE];
#pragma unroll
    for (int e = 0; e < NE; ++e) acc[e] = 0.f;

    for (int d = lane; d < DD; d += 32) {
        const float xv = xr[d];
        const v8f r = *(const v8f*)(rw + (size_t)d * NE);
#pragma unroll
        for (int e = 0; e < NE; ++e) acc[e] += xv * r[e];
    }
#pragma unroll
    for (int off = 16; off > 0; off >>= 1) {
#pragma unroll
        for (int e = 0; e < NE; ++e) acc[e] += __shfl_xor(acc[e], off, 32);
    }
    if (lane == 0) {
        float s[NE];
#pragma unroll
        for (int e = 0; e < NE; ++e) s[e] = 1.f / (1.f + __expf(-acc[e]));
        int i1 = 0; float v1 = s[0];
#pragma unroll
        for (int e = 1; e < NE; ++e) if (s[e] > v1) { v1 = s[e]; i1 = e; }
        int i2 = -1; float v2 = -1e30f;
#pragma unroll
        for (int e = 0; e < NE; ++e)
            if (e != i1 && s[e] > v2) { v2 = s[e]; i2 = e; }
        float* gt = gates + (size_t)t * NXP;
#pragma unroll
        for (int e = 0; e < NE; ++e)
            gt[e] = (e == i1) ? v1 : ((e == i2) ? v2 : 0.f);
        gt[NE] = 1.0f;   // shared expert
    }
}

// ---------------- Fused expert MLP: gate*x @ w13^T -> SwiGLU -> @ w2^T --------
__global__ __launch_bounds__(256)
void moe_ffn_kernel(const float* __restrict__ x, const float* __restrict__ gates,
                    const float* __restrict__ shared_w13, const float* __restrict__ shared_w2,
                    const float* __restrict__ routed_w13, const float* __restrict__ routed_w2,
                    float* __restrict__ out) {
    const int blk = blockIdx.x;
    const int expert = blk >> 7;        // slow index: same-expert blocks adjacent (L2 reuse)
    const int tile = blk & 127;
    const int t0 = tile * TM;

    const float* w13;
    const float* w2;
    if (expert == NE) { w13 = shared_w13; w2 = shared_w2; }
    else {
        w13 = routed_w13 + (size_t)expert * (2 * FF) * DD;
        w2  = routed_w2  + (size_t)expert * DD * FF;
    }

    const int lane = threadIdx.x & 31;
    const int wave = threadIdx.x >> 5;

    extern __shared__ unsigned short smem[];
    unsigned short* sX   = smem;               // [TM][LDP] gated bf16 x tile (~128.5 KB)
    unsigned short* sAct = smem + TM * LDP;    // [TM][LDP] swiglu act tile   (~128.5 KB)

    // ---- Stage gated bf16 X tile once (cooperative, coalesced) ----
    {
        const int col = threadIdx.x * 8;       // 256 threads x 8 floats = 2048 cols
#pragma unroll 4
        for (int r = 0; r < TM; ++r) {
            const float g = gates[(size_t)(t0 + r) * NXP + expert];
            const v8f v = *(const v8f*)(x + (size_t)(t0 + r) * DD + col) * g;
            BF8Cvt cv;
            cv.b = __builtin_convertvector(v, v8bf);
            *(uint4*)(sX + r * LDP + col) = cv.q;
        }
    }
    __syncthreads();

    // ---- Phase A: h = Xg @ w13^T, act = swiglu(h0,h1) -> LDS ----
    for (int fc = wave; fc < FF / 16; fc += 8) {
        const int fb = fc * 16;
        v8f c0a = {0.f,0.f,0.f,0.f,0.f,0.f,0.f,0.f};
        v8f c0b = {0.f,0.f,0.f,0.f,0.f,0.f,0.f,0.f};
        v8f c1a = {0.f,0.f,0.f,0.f,0.f,0.f,0.f,0.f};
        v8f c1b = {0.f,0.f,0.f,0.f,0.f,0.f,0.f,0.f};
#pragma unroll 2
        for (int k = 0; k < DD; k += 32) {
            const v16bf a0 = load_a_lds(sX,            LDP, k, lane);
            const v16bf a1 = load_a_lds(sX + 16 * LDP, LDP, k, lane);
            const v16bf b0 = load_b_global(w13, DD, fb, k, lane);
            const v16bf b1 = load_b_global(w13 + (size_t)FF * DD, DD, fb, k, lane);
            c0a = wmma_bf16(a0, b0, c0a);
            c0b = wmma_bf16(a1, b0, c0b);
            c1a = wmma_bf16(a0, b1, c1a);
            c1b = wmma_bf16(a1, b1, c1b);
        }
        const int n = lane & 15, mo = (lane >> 4) * 8;
#pragma unroll
        for (int r = 0; r < 8; ++r) {
            const float h0a = c0a[r], h1a = c1a[r];
            const float h0b = c0b[r], h1b = c1b[r];
            sAct[(mo + r) * LDP + fb + n]      = f2bf(h0a * h1a / (1.f + __expf(-h0a)));
            sAct[(16 + mo + r) * LDP + fb + n] = f2bf(h0b * h1b / (1.f + __expf(-h0b)));
        }
    }
    __syncthreads();

    // ---- Phase B: z = act @ w2^T, atomically accumulate into out ----
    for (int dc = wave; dc < DD / 16; dc += 8) {
        const int db = dc * 16;
        v8f ca = {0.f,0.f,0.f,0.f,0.f,0.f,0.f,0.f};
        v8f cb = {0.f,0.f,0.f,0.f,0.f,0.f,0.f,0.f};
#pragma unroll 2
        for (int k = 0; k < FF; k += 32) {
            const v16bf a0 = load_a_lds(sAct,            LDP, k, lane);
            const v16bf a1 = load_a_lds(sAct + 16 * LDP, LDP, k, lane);
            const v16bf b  = load_b_global(w2, FF, db, k, lane);
            ca = wmma_bf16(a0, b, ca);
            cb = wmma_bf16(a1, b, cb);
        }
        const int n = lane & 15, mo = (lane >> 4) * 8;
#pragma unroll
        for (int r = 0; r < 8; ++r) {
            atomicAdd(&out[(size_t)(t0 + mo + r) * DD + db + n],      ca[r]);
            atomicAdd(&out[(size_t)(t0 + 16 + mo + r) * DD + db + n], cb[r]);
        }
    }
}

extern "C" void kernel_launch(void* const* d_in, const int* in_sizes, int n_in,
                              void* d_out, int out_size, void* d_ws, size_t ws_size,
                              hipStream_t stream) {
    (void)in_sizes; (void)n_in; (void)out_size; (void)ws_size;
    const float* x          = (const float*)d_in[0];
    const float* router_DE  = (const float*)d_in[1];
    const float* shared_w13 = (const float*)d_in[2];
    const float* shared_w2  = (const float*)d_in[3];
    const float* routed_w13 = (const float*)d_in[4];
    const float* routed_w2  = (const float*)d_in[5];
    float* out   = (float*)d_out;
    float* gates = (float*)d_ws;   // TT*9 floats = 148 KB

    hipMemsetAsync(out, 0, (size_t)TT * DD * sizeof(float), stream);
    router_kernel<<<TT / 8, 256, 0, stream>>>(x, router_DE, gates);
    const size_t ldsBytes = (size_t)2 * TM * LDP * sizeof(unsigned short);  // ~257 KB
    moe_ffn_kernel<<<NXP * (TT / TM), 256, ldsBytes, stream>>>(
        x, gates, shared_w13, shared_w2, routed_w13, routed_w2, out);
}
